// DelayLMLIFLayer_56272661512528
// MI455X (gfx1250) — matrix operand order repaired and checked by reading, not compiled, not run on hardware
//
#include <hip/hip_runtime.h>
#include <hip/hip_bf16.h>

// ---------------- problem constants ----------------
#define T_   1024
#define B_   32
#define J_   128
#define I_   512
#define KD_  16
#define R_   (J_ * KD_)          // 2048 reduction length (r = k*J + j)
#define NROW (T_ * B_)           // 32768 GEMM rows

typedef __attribute__((ext_vector_type(16))) __bf16 v16bf;
typedef __attribute__((ext_vector_type(8)))  __bf16 v8bf;
typedef __attribute__((ext_vector_type(8)))  float  v8f;

// ---------------- workspace layout (bytes) ----------------
#define WS_XH     0              // bf16 x[T][B][J]           (8 MiB)
#define WS_W      (8u  << 20)    // bf16 W[I][R]              (2 MiB)
#define WS_STATS  (10u << 20)    // f32 sum/sumsq/scale/shift (8 KiB)
#define WS_Y      (16u << 20)    // f32 y[T*B][I]             (64 MiB)

// LDS B-slab row stride (bf16 elems): 72 = 144B = 36 dwords.
// 16B-aligned rows; lane stride 36 dwords is bank-conflict-free over 16 lanes.
#define LBS 72

// ============ 1) convert x f32 -> bf16 (row-major, j contiguous) ============
__global__ void prep_x(const float* __restrict__ x, __bf16* __restrict__ xh) {
  int idx = blockIdx.x * blockDim.x + threadIdx.x;   // T*B*J = 4M threads
  xh[idx] = (__bf16)x[idx];
}

// ===== 2) DCLS gaussian kernel: (I,J) w,P -> bf16 W[i][k*J+j] ==============
__global__ void prep_weights(const float* __restrict__ w,
                             const float* __restrict__ P,
                             __bf16* __restrict__ W) {
  int idx = blockIdx.x * blockDim.x + threadIdx.x;   // I*J = 65536
  if (idx >= I_ * J_) return;
  const int i = idx / J_, j = idx % J_;
  const float c = P[idx] + (float)(KD_ / 2);
  float g[KD_];
  float s = 0.f;
#pragma unroll
  for (int k = 0; k < KD_; ++k) {
    float d = 2.0f * ((float)k - c);                 // (k-c)/SIG, SIG=0.5
    float e = __expf(-0.5f * d * d);
    g[k] = e; s += e;
  }
  const float wn = w[idx] / (s + 1e-7f);
#pragma unroll
  for (int k = 0; k < KD_; ++k)
    W[(size_t)i * R_ + k * J_ + j] = (__bf16)(g[k] * wn);
}

// ============ 3) WMMA GEMM: y[n][i] = sum_r A[n][r] * W[i][r] ===============
// Block = 256 threads = 8 waves. Block tile: 128 rows (M) x 64 channels (N).
// Waves stack along M; the shared 64-channel B panel is staged through
// double-buffered LDS in 64-wide K-slabs (one delay tap k, 64 contiguous j,
// i.e. two 32-K WMMA chunks) -> 8 wmma per barrier.
// Hot path (t >= 15, 252/256 row groups): manually unrolled x2 with
// ping-ponged A registers (no rotation copies) and no causal guard.
__global__ void __launch_bounds__(256, 2)
gemm_wmma(const __bf16* __restrict__ xh, const __bf16* __restrict__ W,
          float* __restrict__ y) {
  __shared__ __bf16 lsB[2][64 * LBS];        // 2 x 9KB

  const int nGroup = blockIdx.x & 7;         // 8 channel groups of 64
  const int mGroup = blockIdx.x >> 3;        // 256 row groups of 128
  const int tid    = threadIdx.x;
  // Wave index is wave-uniform: force it scalar so t / trow / the causal
  // guard live on the SALU path.
  const int wave = __builtin_amdgcn_readfirstlane(tid >> 5);
  const int lane = tid & 31;

  const int i0      = nGroup << 6;
  const int rowBase = (mGroup << 7) + (wave << 4);   // wave's 16-row tile
  const int t       = rowBase >> 5;          // 16 rows share one t (B=32)
  const int b0      = rowBase & 31;          // 0 or 16

  const int lrow = lane & 15;                // A row (m) / B column (n)
  const int hiH  = lane >> 4;                // half-wave select
  // A 16x32 layout (matches B): lanes<16 hold K 0..15, lanes>=16 hold K 16..31
  const int aOff0 = hiH ? 16 : 0;
  const int aOff1 = aOff0 + 8;
  // bf16 B 32x16 layout used here (matches A): lanes<16 hold K 0..15, lanes>=16 hold K 16..31
  const int bOff = hiH ? 16 : 0;

  // B global->LDS staging: each thread moves 16 bf16 (32B) per 64-K slab.
  const int sCh  = tid >> 2;                 // 0..63 channel within panel
  const int sSeg = (tid & 3) << 4;           // 0,16,32,48 within slab
  const __bf16* wgp = W + (size_t)(i0 + sCh) * R_ + sSeg;
  __bf16* lsw0 = &lsB[0][sCh * LBS + sSeg];
  __bf16* lsw1 = &lsB[1][sCh * LBS + sSeg];

  const v16bf vzero = {};
  v8f acc[4] = {{}, {}, {}, {}};

  auto stage = [&](int s, __bf16* dst) {
    const __bf16* wp = wgp + (s << 6);       // slab s covers r in [64s,64s+64)
    v8bf w0 = *(const v8bf*)(wp);
    v8bf w1 = *(const v8bf*)(wp + 8);
    *(v8bf*)(dst)     = w0;
    *(v8bf*)(dst + 8) = w1;
  };

  // Unguarded A fragment loads for slab s (requires trow >= 0).
  auto loadA2f = [&](int s, v16bf& A0, v16bf& A1) {
    const int k    = s >> 1;
    const int j0   = (s & 1) << 6;
    const int trow = t + k - (KD_ - 1);
    const __bf16* ap = xh + ((size_t)trow * B_ + b0 + lrow) * J_ + j0;
    v8bf l0 = *(const v8bf*)(ap + aOff0);
    v8bf h0 = *(const v8bf*)(ap + aOff1);
    A0 = __builtin_shufflevector(l0, h0, 0,1,2,3,4,5,6,7,
                                         8,9,10,11,12,13,14,15);
    v8bf l1 = *(const v8bf*)(ap + 32 + aOff0);
    v8bf h1 = *(const v8bf*)(ap + 32 + aOff1);
    A1 = __builtin_shufflevector(l1, h1, 0,1,2,3,4,5,6,7,
                                         8,9,10,11,12,13,14,15);
  };

  auto ldsfrag = [&](const __bf16* bp) -> v16bf {
    v8bf lo = *(const v8bf*)(bp);            // ds_load_b128
    v8bf hi = *(const v8bf*)(bp + 8);
    return __builtin_shufflevector(lo, hi, 0,1,2,3,4,5,6,7,
                                           8,9,10,11,12,13,14,15);
  };

  // 8 wmma on one staged slab (two 32-K chunks x 4 channel tiles).
  auto mmaGroup = [&](const v16bf& A0, const v16bf& A1, const __bf16* base) {
#pragma unroll
    for (int tt = 0; tt < 4; ++tt) {
      const __bf16* bp = base + (tt * 16 + lrow) * LBS + bOff;
      acc[tt] = __builtin_amdgcn_wmma_f32_16x16x32_bf16(
          false, A0, false, ldsfrag(bp), (short)0, acc[tt], false, false);
      acc[tt] = __builtin_amdgcn_wmma_f32_16x16x32_bf16(
          false, A1, false, ldsfrag(bp + 32), (short)0, acc[tt], false, false);
    }
  };

  if (t >= KD_ - 1) {
    // ---------------- hot path: every slab valid, no guards ----------------
    v16bf pa0, pa1, pb0, pb1;                // ping-pong A register sets
    stage(0, lsw0);
    loadA2f(0, pa0, pa1);
    __syncthreads();
    for (int s = 0; s < 32; s += 2) {
      // slab s (even, buf0): prefetch slab s+1 into buf1 + B-regs
      stage(s + 1, lsw1);
      loadA2f(s + 1, pb0, pb1);
      mmaGroup(pa0, pa1, &lsB[0][0]);
      __syncthreads();
      // slab s+1 (odd, buf1): prefetch slab s+2 into buf0 + A-regs
      if (s + 2 < 32) {
        stage(s + 2, lsw0);
        loadA2f(s + 2, pa0, pa1);
      }
      mmaGroup(pb0, pb1, &lsB[1][0]);
      __syncthreads();
    }
  } else {
    // ------------- edge path (t < 15): guarded, same barrier count ---------
    auto loadA2 = [&](int s, v16bf& A0, v16bf& A1) {
      A0 = vzero; A1 = vzero;                // zeros in the padding region
      if (t + (s >> 1) - (KD_ - 1) >= 0) loadA2f(s, A0, A1);
    };
    v16bf a0, a1;
    stage(0, lsw0);
    loadA2(0, a0, a1);
    __syncthreads();
    for (int s = 0; s < 32; ++s) {
      v16bf a0n = vzero, a1n = vzero;
      if (s + 1 < 32) {
        stage(s + 1, (s & 1) ? lsw0 : lsw1);
        loadA2(s + 1, a0n, a1n);
      }
      mmaGroup(a0, a1, &lsB[s & 1][0]);
      a0 = a0n; a1 = a1n;
      __syncthreads();
    }
  }

  // C/D layout: VGPR v -> M = v + 8*hiH, N = lrow
#pragma unroll
  for (int v = 0; v < 8; ++v) {
    const int m = v + hiH * 8;
    float* yp = y + (size_t)(rowBase + m) * I_ + i0 + lrow;
    yp[0]  = acc[0][v];
    yp[16] = acc[1][v];
    yp[32] = acc[2][v];
    yp[48] = acc[3][v];
  }
}

// ============ 4) BN stats: per-channel sum / sumsq over 32768 rows =========
__global__ void bn_stats(const float* __restrict__ y,
                         float* __restrict__ sum, float* __restrict__ sumsq) {
  const int blk = blockIdx.x;                // 128 blocks x 256 rows
  const int tid = threadIdx.x;
#pragma unroll
  for (int ci = 0; ci < 2; ++ci) {
    const int i = tid + ci * 256;
    const float* p = y + (size_t)blk * 256 * I_ + i;
    float s = 0.f, ss = 0.f;
    for (int n = 0; n < 256; ++n) {
      float v = p[(size_t)n * I_];
      s += v; ss += v * v;
    }
    atomicAdd(&sum[i], s);
    atomicAdd(&sumsq[i], ss);
  }
}

__global__ void bn_finalize(const float* __restrict__ sum,
                            const float* __restrict__ sumsq,
                            const float* __restrict__ gamma,
                            const float* __restrict__ bbeta,
                            float* __restrict__ scale,
                            float* __restrict__ shift) {
  int i = blockIdx.x * blockDim.x + threadIdx.x;
  if (i >= I_) return;
  const float invN = 1.0f / (float)NROW;
  const float mean = sum[i] * invN;
  const float var  = sumsq[i] * invN - mean * mean;   // ddof=0, matches jnp.var
  const float sc   = gamma[i] * rsqrtf(var + 1e-5f);
  scale[i] = sc;
  shift[i] = bbeta[i] - mean * sc;
}

// ============ 5) soft-reset LIF scan (serial in t, parallel over B*I) ======
__global__ void lif_scan(const float* __restrict__ y,
                         const float* __restrict__ scale,
                         const float* __restrict__ shift,
                         const float* __restrict__ beta,
                         const float* __restrict__ U0,
                         float* __restrict__ out) {
  const int idx = blockIdx.x * blockDim.x + threadIdx.x;  // B*I = 16384
  const int b = idx / I_, i = idx % I_;
  const float be = beta[i], ob = 1.0f - be;
  const float sc = scale[i], sh = shift[i];
  float U = U0[(size_t)b * I_ + i];
  float S = 0.0f;
#pragma unroll 4
  for (int t = 0; t < T_; ++t) {
    const size_t off = ((size_t)t * B_ + b) * I_ + i;     // i-fastest: coalesced
    const float v = y[off] * sc + sh;                     // y sits in 192MB L2
    U = be * (U - S) + ob * v;                            // THETA = 1
    S = (U >= 1.0f) ? 1.0f : 0.0f;                        // atan_spike fwd
    __builtin_nontemporal_store(S, &out[off]);            // don't evict y
  }
}

// ============================ launcher =====================================
extern "C" void kernel_launch(void* const* d_in, const int* in_sizes, int n_in,
                              void* d_out, int out_size, void* d_ws, size_t ws_size,
                              hipStream_t stream) {
  const float* x       = (const float*)d_in[0];   // (T,B,J)
  const float* delay_w = (const float*)d_in[1];   // (I,J)
  const float* delay_P = (const float*)d_in[2];   // (I,J)
  const float* beta    = (const float*)d_in[3];   // (I,)
  const float* gamma   = (const float*)d_in[4];   // (I,)
  const float* bbeta   = (const float*)d_in[5];   // (I,)
  const float* U0      = (const float*)d_in[6];   // (B,I)
  float* out = (float*)d_out;                     // (T,B,I)

  char* ws = (char*)d_ws;
  __bf16* xh   = (__bf16*)(ws + WS_XH);
  __bf16* W    = (__bf16*)(ws + WS_W);
  float*  sum  = (float*)(ws + WS_STATS);
  float*  ssq  = sum + I_;
  float*  scl  = ssq + I_;
  float*  shf  = scl + I_;
  float*  y    = (float*)(ws + WS_Y);

  // 1) x -> bf16
  prep_x<<<(T_ * B_ * J_) / 256, 256, 0, stream>>>(x, xh);
  // 2) gaussian delay kernel -> bf16 W[i][k*J+j]
  prep_weights<<<(I_ * J_) / 256, 256, 0, stream>>>(delay_w, delay_P, W);
  // 3) zero BN accumulators (must happen every call: deterministic replay)
  hipMemsetAsync(sum, 0, 2 * I_ * sizeof(float), stream);
  // 4) WMMA conv-as-GEMM: 256 row-groups x 8 channel-groups
  gemm_wmma<<<(NROW / 128) * (I_ / 64), 256, 0, stream>>>(xh, W, y);
  // 5) BN statistics + fold into scale/shift
  bn_stats<<<NROW / 256, 256, 0, stream>>>(y, sum, ssq);
  bn_finalize<<<I_ / 256, 256, 0, stream>>>(sum, ssq, gamma, bbeta, scl, shf);
  // 6) LIF scan
  lif_scan<<<(B_ * I_) / 256, 256, 0, stream>>>(y, scl, shf, beta, U0, out);
}
